// MLPModelPredictor_52604759441782
// MI455X (gfx1250) — matrix operand order repaired and checked by reading, not compiled, not run on hardware
//
#include <hip/hip_runtime.h>

typedef _Float16 half_t;
typedef _Float16 v16h __attribute__((ext_vector_type(16)));
typedef _Float16 v8h  __attribute__((ext_vector_type(8)));
typedef _Float16 v4h  __attribute__((ext_vector_type(4)));
typedef float    v8f  __attribute__((ext_vector_type(8)));
typedef int      v4i  __attribute__((ext_vector_type(4)));

#define D_S    2048
#define D_B    16
#define D_F    100
#define D_E    512
#define D_H    512
#define D_NOUT 10
#define D_DEC  1024
#define D_FF   2048
#define D_NH   8
#define D_DH   64
#define D_SEP  1024
#define D_TTOK (D_SEP * D_B)
#define D_EVAL (D_S - D_SEP)

// gfx1250 async Global->LDS copy path for f16 A tiles (ASYNCcnt-tracked DMA).
#define ASYNC_TRY 1
#if ASYNC_TRY && defined(__has_builtin)
#if __has_builtin(__builtin_amdgcn_global_load_async_to_lds_b128)
#define HAVE_ASYNC_LDS 1
#endif
#endif

#if defined(HAVE_ASYNC_LDS)
typedef __attribute__((address_space(3))) void* las_void;
typedef __attribute__((address_space(3))) v4i*  las_v4i;
__device__ __forceinline__ void wait_asynccnt0() {
#if __has_builtin(__builtin_amdgcn_s_wait_asynccnt)
    __builtin_amdgcn_s_wait_asynccnt(0);
#else
    asm volatile("s_wait_asynccnt 0" ::: "memory");
#endif
}
#endif

// ---------------------------------------------------------------------------
// Generic WMMA GEMM: C = act(A @ B + bias + y-rank1).
// Block: 256 threads = 8 waves; tile 64(M) x 128(N); K step 32.
// B tile stored TRANSPOSED in LDS -> both fragments are 2x ds_load_b128.
// Dtypes are template parameters so tile-load codegen is branch-free.
// ---------------------------------------------------------------------------
struct GemmParams {
    const void* A;   long lda;  long aBatch; int sanA;
    const void* Bw;  long ldb;  long bBatch;
    const float* bias;  long biasBatch;
    const float* yvec;  const float* yW;  const float* yB;   // out += y[row]*yW[col]+yB[col]
    float*  outF; long ldoF; long outFBatch;
    half_t* outH; long ldoH; long outHBatch;
    int M, N, K, relu;
};

#define TM 64
#define TN 128
#define TK 32
#define AS_STRIDE 48   // 96B rows -> 16B-aligned v8h fragment loads
#define BT_STRIDE 40   // 80B rows -> 16B-aligned v8h fragment loads (80 = 5*16)

template <int AF16, int BF16>
__global__ __launch_bounds__(256) void gemm_wmma_kernel(GemmParams p) {
    __shared__ __align__(16) half_t As[TM * AS_STRIDE];   // [m][k]
    __shared__ __align__(16) half_t BsT[TN * BT_STRIDE];  // [n][k] (transposed)

    const int tid  = threadIdx.x;
    const int lane = tid & 31;
    const int wave = tid >> 5;
    const int wr   = wave & 3;      // wave row  (4 x 16 = 64 M)
    const int wc   = wave >> 2;     // wave col  (2 x 64 = 128 N)
    const int lo16 = lane & 15;
    const int hiH  = (lane >= 16);

    const int m0 = blockIdx.y * TM;
    const int n0 = blockIdx.x * TN;
    const long zb = blockIdx.z;

    const float*  Af = (const float*)p.A;
    const half_t* Ah = (const half_t*)p.A;
    const float*  Bf = (const float*)p.Bw;
    const half_t* Bh = (const half_t*)p.Bw;
    const long aOff = zb * p.aBatch;
    const long bOff = zb * p.bBatch;

    // vector fast-path alignment (uniform per launch)
    const bool aFast = AF16 ? ((p.lda & 7) == 0 && (p.aBatch & 7) == 0)
                            : ((p.lda & 3) == 0 && (p.aBatch & 3) == 0);
    const bool bFast = ((p.ldb & 3) == 0) && ((p.bBatch & 3) == 0);

    v8f acc[4];
#pragma unroll
    for (int c = 0; c < 4; c++)
#pragma unroll
        for (int r = 0; r < 8; r++) acc[c][r] = 0.f;

    const int aRow = tid >> 2;           // 64 rows, 8 k-elems each
    const int aCol = (tid & 3) * 8;
    const int n4 = (tid & 31) * 4;       // 4 n-columns per thread
    const int kq = (tid >> 5) * 4;       // 4 k-rows per thread

    for (int kk = 0; kk < p.K; kk += TK) {
        __syncthreads();
        {   // ---- A tile 64x32 -> LDS row-major
            const int grow = m0 + aRow;
            const long abase = aOff + (long)grow * p.lda + kk + aCol;
            const bool inFast = (grow < p.M) && aFast && (kk + aCol + 8 <= p.K);
            bool stored = false;
#if defined(HAVE_ASYNC_LDS)
            if (AF16 && inFast) {
                // async DMA: global (b128) -> LDS, no VGPR round-trip; ASYNCcnt
                __builtin_amdgcn_global_load_async_to_lds_b128(
                    (v4i*)&Ah[abase],
                    (las_v4i)(las_void)&As[aRow * AS_STRIDE + aCol], 0, 0);
                stored = true;
            }
#endif
            if (!stored) {
                float tmp[8];
                if (inFast) {
                    if (AF16) {
                        v8h t = *(const v8h*)&Ah[abase];
#pragma unroll
                        for (int j = 0; j < 8; j++) tmp[j] = (float)t[j];
                    } else {
                        float4 f0 = *(const float4*)&Af[abase];
                        float4 f1 = *(const float4*)&Af[abase + 4];
                        tmp[0] = f0.x; tmp[1] = f0.y; tmp[2] = f0.z; tmp[3] = f0.w;
                        tmp[4] = f1.x; tmp[5] = f1.y; tmp[6] = f1.z; tmp[7] = f1.w;
                    }
                } else {
#pragma unroll
                    for (int j = 0; j < 8; j++) {
                        int gk = kk + aCol + j;
                        tmp[j] = (grow < p.M && gk < p.K)
                                     ? (AF16 ? (float)Ah[abase + j] : Af[abase + j])
                                     : 0.f;
                    }
                }
                v8h av;
#pragma unroll
                for (int j = 0; j < 8; j++) {
                    float v = tmp[j];
                    if (!AF16 && p.sanA && !(v == v)) v = 0.f;   // nan_to_num
                    av[j] = (half_t)v;
                }
                *(v8h*)&As[aRow * AS_STRIDE + aCol] = av;
            }
        }
        {   // ---- B tile 32x128: coalesced global rows, register 4x4 transpose,
            //      b64 column stores into BsT[n][k]
            const int gk0 = kk + kq;
            half_t bv[4][4];                                  // [dk][dn]
            const bool fast = bFast && (gk0 + 4 <= p.K) && (n0 + n4 + 4 <= p.N);
            if (fast) {
#pragma unroll
                for (int dk = 0; dk < 4; dk++) {
                    const long bb = bOff + (long)(gk0 + dk) * p.ldb + n0 + n4;
                    if (BF16) {
                        v4h t = *(const v4h*)&Bh[bb];
#pragma unroll
                        for (int dn = 0; dn < 4; dn++) bv[dk][dn] = t[dn];
                    } else {
                        float4 t = *(const float4*)&Bf[bb];
                        bv[dk][0] = (half_t)t.x; bv[dk][1] = (half_t)t.y;
                        bv[dk][2] = (half_t)t.z; bv[dk][3] = (half_t)t.w;
                    }
                }
                // prefetch next k-tile of the B stream (global_prefetch_b8)
                if (kk + TK < p.K) {
                    const char* pf = BF16
                        ? (const char*)&Bh[bOff + (long)(gk0 + TK) * p.ldb + n0 + n4]
                        : (const char*)&Bf[bOff + (long)(gk0 + TK) * p.ldb + n0 + n4];
                    __builtin_prefetch(pf, 0, 1);
                }
            } else {
#pragma unroll
                for (int dk = 0; dk < 4; dk++)
#pragma unroll
                    for (int dn = 0; dn < 4; dn++) {
                        int gk = gk0 + dk, gn = n0 + n4 + dn;
                        float v = 0.f;
                        if (gk < p.K && gn < p.N)
                            v = BF16 ? (float)Bh[bOff + (long)gk * p.ldb + gn]
                                     : Bf[bOff + (long)gk * p.ldb + gn];
                        bv[dk][dn] = (half_t)v;
                    }
            }
#pragma unroll
            for (int dn = 0; dn < 4; dn++) {
                v4h st;
                st[0] = bv[0][dn]; st[1] = bv[1][dn];
                st[2] = bv[2][dn]; st[3] = bv[3][dn];
                *(v4h*)&BsT[(n4 + dn) * BT_STRIDE + kq] = st;
            }
        }
#if defined(HAVE_ASYNC_LDS)
        if (AF16 && aFast) wait_asynccnt0();   // uniform: async A writes visible
#endif
        __syncthreads();

        // A fragment 16x32 (lanes0-15: K{0-7,16-23}; lanes16-31: K{8-15,24-31})
        v16h a;
        {
            const int klo  = hiH ? 8 : 0;
            const int arow = wr * 16 + lo16;
            v8h alo = *(const v8h*)&As[arow * AS_STRIDE + klo];
            v8h ahi = *(const v8h*)&As[arow * AS_STRIDE + 16 + klo];
            a = __builtin_shufflevector(alo, ahi, 0,1,2,3,4,5,6,7,8,9,10,11,12,13,14,15);
        }
        const int kb = hiH ? 8 : 0;
#pragma unroll
        for (int c = 0; c < 4; c++) {
            const int col = wc * 64 + c * 16 + lo16;
            v8h bl = *(const v8h*)&BsT[col * BT_STRIDE + kb];
            v8h bh = *(const v8h*)&BsT[col * BT_STRIDE + kb + 16];
            v16h bfr = __builtin_shufflevector(bl, bh, 0,1,2,3,4,5,6,7,8,9,10,11,12,13,14,15);
            acc[c] = __builtin_amdgcn_wmma_f32_16x16x32_f16(
                false, a, false, bfr, (short)0, acc[c], false, false);
        }
    }

    // Epilogue (C layout: VGPR r -> row r / r+8, col = lane&15)
    const bool fullTile = (m0 + TM <= p.M) && (n0 + TN <= p.N);
    if (fullTile) {
#pragma unroll
        for (int c = 0; c < 4; c++) {
            const int gcol = n0 + wc * 64 + c * 16 + lo16;
            float badd = p.bias ? p.bias[zb * p.biasBatch + gcol] : 0.f;
#pragma unroll
            for (int r = 0; r < 8; r++) {
                const int grow = m0 + wr * 16 + r + (hiH ? 8 : 0);
                float v = acc[c][r] + badd;
                if (p.yvec) v += p.yvec[grow] * p.yW[gcol] + p.yB[gcol];
                if (p.relu) v = fmaxf(v, 0.f);
                if (p.outF) p.outF[zb * p.outFBatch + (long)grow * p.ldoF + gcol] = v;
                if (p.outH) p.outH[zb * p.outHBatch + (long)grow * p.ldoH + gcol] = (half_t)v;
            }
        }
    } else {
#pragma unroll
        for (int c = 0; c < 4; c++) {
            const int gcol = n0 + wc * 64 + c * 16 + lo16;
            if (gcol >= p.N) continue;
            float badd = p.bias ? p.bias[zb * p.biasBatch + gcol] : 0.f;
#pragma unroll
            for (int r = 0; r < 8; r++) {
                const int grow = m0 + wr * 16 + r + (hiH ? 8 : 0);
                if (grow >= p.M) continue;
                float v = acc[c][r] + badd;
                if (p.yvec) v += p.yvec[grow] * p.yW[gcol] + p.yB[gcol];
                if (p.relu) v = fmaxf(v, 0.f);
                if (p.outF) p.outF[zb * p.outFBatch + (long)grow * p.ldoF + gcol] = v;
                if (p.outH) p.outH[zb * p.outHBatch + (long)grow * p.ldoH + gcol] = (half_t)v;
            }
        }
    }
}

// ---------------------------------------------------------------------------
// Flash attention: one wave per (b, h, 16-query tile). Q in registers as
// A-frags, K read directly as B-frags (two 16B loads per lane, 32B apart), V staged
// TRANSPOSED in LDS (register 4x16 transpose) so V-fragments are 2x b128,
// online softmax with 16-lane row reductions, P via LDS C->A relayout.
// ---------------------------------------------------------------------------
#define VT_STRIDE 40   // vT[d][k] rows: 80B, 16B-aligned

__global__ __launch_bounds__(32) void attn_kernel(const half_t* __restrict__ qh,
                                                  const half_t* __restrict__ kh,
                                                  const half_t* __restrict__ vh,
                                                  half_t* __restrict__ aoh) {
    __shared__ __align__(16) half_t vT[64 * VT_STRIDE];   // [d][k]
    __shared__ __align__(16) half_t ptile[16 * 32];

    const int lane = threadIdx.x;
    const int bh = blockIdx.y;
    const int b  = bh / D_NH;
    const int h  = bh % D_NH;
    const int q0 = blockIdx.x * 16;
    const int colE = h * D_DH;
    const int lo16 = lane & 15;
    const int hiH  = (lane >= 16);
    const int kb   = hiH ? 8 : 0;
    const int klo  = hiH ? 8 : 0;

    v16h qa[2];
    {
        long base = ((long)(q0 + lo16) * D_B + b) * D_E + colE;
#pragma unroll
        for (int ka = 0; ka < 2; ka++) {
            v8h alo = *(const v8h*)&qh[base + ka * 32 + klo];
            v8h ahi = *(const v8h*)&qh[base + ka * 32 + 16 + klo];
            qa[ka] = __builtin_shufflevector(alo, ahi, 0,1,2,3,4,5,6,7,8,9,10,11,12,13,14,15);
        }
    }

    float mst[8], lst[8];
    v8f o[4];
#pragma unroll
    for (int r = 0; r < 8; r++) { mst[r] = -1e30f; lst[r] = 0.f; }
#pragma unroll
    for (int c = 0; c < 4; c++)
#pragma unroll
        for (int r = 0; r < 8; r++) o[c][r] = 0.f;

    const int d4  = lo16 * 4;           // 4 d-columns per lane for V staging
    const int kq  = hiH ? 16 : 0;       // 16 keys per lane group

    for (int kt = 0; kt < D_SEP / 32; kt++) {
        const int key0 = kt * 32;
        __syncthreads();
        {   // stage V tile 32x64 transposed: coalesced b64 rows -> b128 columns
            half_t vals[16][4];
#pragma unroll
            for (int dk = 0; dk < 16; dk++) {
                v4h t = *(const v4h*)&vh[((long)(key0 + kq + dk) * D_B + b) * D_E + colE + d4];
#pragma unroll
                for (int dd = 0; dd < 4; dd++) vals[dk][dd] = t[dd];
            }
#pragma unroll
            for (int dd = 0; dd < 4; dd++) {
                v8h lo, hi;
#pragma unroll
                for (int dk = 0; dk < 8; dk++) { lo[dk] = vals[dk][dd]; hi[dk] = vals[8 + dk][dd]; }
                *(v8h*)&vT[(d4 + dd) * VT_STRIDE + kq]     = lo;
                *(v8h*)&vT[(d4 + dd) * VT_STRIDE + kq + 8] = hi;
            }
        }
        // S = Q(16x64) x K^T(64x32): two key-column frags, two k-steps
        v8f s0, s1;
#pragma unroll
        for (int r = 0; r < 8; r++) { s0[r] = 0.f; s1[r] = 0.f; }
#pragma unroll
        for (int ka = 0; ka < 2; ka++) {
            long basea = ((long)(key0 + lo16) * D_B + b) * D_E + colE + ka * 32;
            long baseb = ((long)(key0 + 16 + lo16) * D_B + b) * D_E + colE + ka * 32;
            v8h k0l = *(const v8h*)&kh[basea + kb];
            v8h k0h = *(const v8h*)&kh[basea + kb + 16];
            v8h k1l = *(const v8h*)&kh[baseb + kb];
            v8h k1h = *(const v8h*)&kh[baseb + kb + 16];
            v16h bf0 = __builtin_shufflevector(k0l, k0h, 0,1,2,3,4,5,6,7,8,9,10,11,12,13,14,15);
            v16h bf1 = __builtin_shufflevector(k1l, k1h, 0,1,2,3,4,5,6,7,8,9,10,11,12,13,14,15);
            s0 = __builtin_amdgcn_wmma_f32_16x16x32_f16(false, qa[ka], false, bf0, (short)0, s0, false, false);
            s1 = __builtin_amdgcn_wmma_f32_16x16x32_f16(false, qa[ka], false, bf1, (short)0, s1, false, false);
        }
        // online softmax over the 32-key tile
        float sc8[8];
#pragma unroll
        for (int r = 0; r < 8; r++) {
            float v0 = s0[r] * 0.125f;   // 1/sqrt(64)
            float v1 = s1[r] * 0.125f;
            float rm = fmaxf(v0, v1);
#pragma unroll
            for (int m = 1; m < 16; m <<= 1) rm = fmaxf(rm, __shfl_xor(rm, m, 32));
            float mnew = fmaxf(mst[r], rm);
            float sc = __expf(mst[r] - mnew);
            float p0 = __expf(v0 - mnew);
            float p1 = __expf(v1 - mnew);
            float rs = p0 + p1;
#pragma unroll
            for (int m = 1; m < 16; m <<= 1) rs += __shfl_xor(rs, m, 32);
            lst[r] = lst[r] * sc + rs;
            mst[r] = mnew;
            sc8[r] = sc;
            int prow = r + (hiH ? 8 : 0);
            ptile[prow * 32 + lo16]      = (half_t)p0;
            ptile[prow * 32 + lo16 + 16] = (half_t)p1;
        }
#pragma unroll
        for (int c = 0; c < 4; c++)
#pragma unroll
            for (int r = 0; r < 8; r++) o[c][r] *= sc8[r];
        __syncthreads();
        // P as A-frag, V as B-frags; O += P(16x32) x V(32x64)
        v16h pa;
        {
            v8h plo = *(const v8h*)&ptile[lo16 * 32 + klo];
            v8h phi = *(const v8h*)&ptile[lo16 * 32 + 16 + klo];
            pa = __builtin_shufflevector(plo, phi, 0,1,2,3,4,5,6,7,8,9,10,11,12,13,14,15);
        }
#pragma unroll
        for (int c = 0; c < 4; c++) {
            const int col = c * 16 + lo16;
            v8h bl = *(const v8h*)&vT[col * VT_STRIDE + kb];
            v8h bh2 = *(const v8h*)&vT[col * VT_STRIDE + kb + 16];
            v16h vb = __builtin_shufflevector(bl, bh2, 0,1,2,3,4,5,6,7,8,9,10,11,12,13,14,15);
            o[c] = __builtin_amdgcn_wmma_f32_16x16x32_f16(false, pa, false, vb, (short)0, o[c], false, false);
        }
    }
#pragma unroll
    for (int c = 0; c < 4; c++) {
        int gcol = colE + c * 16 + lo16;
#pragma unroll
        for (int r = 0; r < 8; r++) {
            int row = q0 + r + (hiH ? 8 : 0);
            aoh[((long)row * D_B + b) * D_E + gcol] = (half_t)(o[c][r] / lst[r]);
        }
    }
}

// ---------------------------------------------------------------------------
// LayerNorm over E=512 of (in1 + in2); one wave per token.
// ---------------------------------------------------------------------------
__global__ __launch_bounds__(256) void ln_kernel(const float* __restrict__ in1,
                                                 const float* __restrict__ in2,
                                                 const float* __restrict__ g,
                                                 const float* __restrict__ bb,
                                                 float* __restrict__ outF,
                                                 half_t* __restrict__ outH, int ntok) {
    const int wave = threadIdx.x >> 5;
    const int lane = threadIdx.x & 31;
    const int tok = blockIdx.x * 8 + wave;
    if (tok >= ntok) return;
    const long base = (long)tok * D_E;
    float v[16], s = 0.f, s2 = 0.f;
#pragma unroll
    for (int i = 0; i < 16; i++) {
        int c = lane + 32 * i;
        v[i] = in1[base + c] + in2[base + c];
        s += v[i];
        s2 += v[i] * v[i];
    }
#pragma unroll
    for (int m = 1; m < 32; m <<= 1) { s += __shfl_xor(s, m, 32); s2 += __shfl_xor(s2, m, 32); }
    float mean = s * (1.f / D_E);
    float var = s2 * (1.f / D_E) - mean * mean;
    float rstd = rsqrtf(var + 1e-5f);
#pragma unroll
    for (int i = 0; i < 16; i++) {
        int c = lane + 32 * i;
        float ov = (v[i] - mean) * rstd * g[c] + bb[c];
        outF[base + c] = ov;
        if (outH) outH[base + c] = (half_t)ov;
    }
}

// pooled[b,e] = mean over T of out[t,b,e]
__global__ void pool_kernel(const float* __restrict__ src, float* __restrict__ pooled) {
    int id = blockIdx.x * blockDim.x + threadIdx.x;
    if (id >= D_B * D_E) return;
    int b = id / D_E, e = id % D_E;
    float acc = 0.f;
    for (int t = 0; t < D_SEP; t++) acc += src[((long)t * D_B + b) * D_E + e];
    pooled[id] = acc * (1.f / D_SEP);
}

// ---------------------------------------------------------------------------
// Host orchestration
// ---------------------------------------------------------------------------
static inline void* ws_take(char* base, size_t& off, size_t bytes) {
    void* p = base + off;
    off += (bytes + 255) & ~(size_t)255;
    return p;
}

static void launch_gemm(const GemmParams& p, int aF16, int bF16, int batch, hipStream_t stream) {
    dim3 grid((p.N + TN - 1) / TN, (p.M + TM - 1) / TM, batch);
    dim3 blk(256);
    if (aF16) {
        if (bF16) gemm_wmma_kernel<1, 1><<<grid, blk, 0, stream>>>(p);
        else      gemm_wmma_kernel<1, 0><<<grid, blk, 0, stream>>>(p);
    } else {
        if (bF16) gemm_wmma_kernel<0, 1><<<grid, blk, 0, stream>>>(p);
        else      gemm_wmma_kernel<0, 0><<<grid, blk, 0, stream>>>(p);
    }
}

extern "C" void kernel_launch(void* const* d_in, const int* in_sizes, int n_in,
                              void* d_out, int out_size, void* d_ws, size_t ws_size,
                              hipStream_t stream) {
    (void)in_sizes; (void)n_in; (void)out_size; (void)ws_size;
    const float* x      = (const float*)d_in[0];
    const float* y      = (const float*)d_in[1];
    const float* enc_W  = (const float*)d_in[2];
    const float* enc_b  = (const float*)d_in[3];
    const float* yenc_W = (const float*)d_in[4];
    const float* yenc_b = (const float*)d_in[5];
    const float* Wq     = (const float*)d_in[6];
    const float* Wk     = (const float*)d_in[7];
    const float* Wv     = (const float*)d_in[8];
    const float* Wo     = (const float*)d_in[9];
    const float* ln1_g  = (const float*)d_in[10];
    const float* ln1_b  = (const float*)d_in[11];
    const float* ln2_g  = (const float*)d_in[12];
    const float* ln2_b  = (const float*)d_in[13];
    const float* ffn_W1 = (const float*)d_in[14];
    const float* ffn_b1 = (const float*)d_in[15];
    const float* ffn_W2 = (const float*)d_in[16];
    const float* ffn_b2 = (const float*)d_in[17];
    const float* dec_W  = (const float*)d_in[18];
    const float* dec_b  = (const float*)d_in[19];
    const float* hw1    = (const float*)d_in[20];
    const float* hb1    = (const float*)d_in[21];
    const float* hw2    = (const float*)d_in[22];
    const float* hb2    = (const float*)d_in[23];
    // d_in[24] = single_eval_pos: fixed 1024 in the reference setup.
    float* out = (float*)d_out;

    char* ws = (char*)d_ws;
    size_t off = 0;
    float*  trainx = (float*) ws_take(ws, off, (size_t)D_TTOK * D_E * 4); // later: LN2 out
    half_t* bufB   = (half_t*)ws_take(ws, off, (size_t)D_TTOK * D_E * 2); // trainx_h -> aoh
    half_t* bufC   = (half_t*)ws_take(ws, off, (size_t)D_TTOK * D_E * 2); // qh -> z_h
    half_t* bufD   = (half_t*)ws_take(ws, off, (size_t)D_TTOK * D_E * 2); // kh -> eval hidden
    half_t* vh     = (half_t*)ws_take(ws, off, (size_t)D_TTOK * D_E * 2);
    float*  scr1   = (float*) ws_take(ws, off, (size_t)D_TTOK * D_E * 4); // attnproj -> ffn_out
    float*  zf     = (float*) ws_take(ws, off, (size_t)D_TTOK * D_E * 4);
    half_t* ffh    = (half_t*)ws_take(ws, off, (size_t)D_TTOK * D_FF * 2);
    float*  pooled = (float*) ws_take(ws, off, (size_t)D_B * D_E * 4);
    half_t* dh_h   = (half_t*)ws_take(ws, off, (size_t)D_B * D_DEC * 2);
    half_t* w1h    = (half_t*)ws_take(ws, off, (size_t)D_B * D_F * D_H * 2);
    float*  b1f    = (float*) ws_take(ws, off, (size_t)D_B * D_H * 4);
    half_t* w2h    = (half_t*)ws_take(ws, off, (size_t)D_B * D_H * D_NOUT * 2);
    float*  b2f    = (float*) ws_take(ws, off, (size_t)D_B * D_NOUT * 4);

    // 1. encoders fused: train_x = x@enc_W + enc_b + y*yenc_W + yenc_b
    { GemmParams p{}; p.A = x; p.lda = D_F;
      p.Bw = enc_W; p.ldb = D_E;
      p.bias = enc_b;
      p.yvec = y; p.yW = yenc_W; p.yB = yenc_b;
      p.outF = trainx; p.ldoF = D_E;
      p.outH = bufB;  p.ldoH = D_E;
      p.M = D_TTOK; p.N = D_E; p.K = D_F;
      launch_gemm(p, 0, 0, 1, stream); }

    // 2-4. Q, K, V projections (f16 out)
    auto proj = [&](const float* W, half_t* dst) {
        GemmParams p{}; p.A = bufB; p.lda = D_E;
        p.Bw = W; p.ldb = D_E;
        p.outH = dst; p.ldoH = D_E;
        p.M = D_TTOK; p.N = D_E; p.K = D_E;
        launch_gemm(p, 1, 0, 1, stream);
    };
    proj(Wq, bufC);
    proj(Wk, bufD);
    proj(Wv, vh);

    // 5. flash attention -> aoh (reuses trainx_h slot)
    attn_kernel<<<dim3(D_SEP / 16, D_B * D_NH), dim3(32), 0, stream>>>(bufC, bufD, vh, bufB);

    // 6. output projection
    { GemmParams p{}; p.A = bufB; p.lda = D_E;
      p.Bw = Wo; p.ldb = D_E;
      p.outF = scr1; p.ldoF = D_E;
      p.M = D_TTOK; p.N = D_E; p.K = D_E;
      launch_gemm(p, 1, 0, 1, stream); }

    // 7. LN1(train_x + attn_out) -> zf (f32) + z_h (f16, reuses qh slot)
    ln_kernel<<<D_TTOK / 8, 256, 0, stream>>>(trainx, scr1, ln1_g, ln1_b, zf, bufC, D_TTOK);

    // 8. FFN1 (+bias, relu) -> f16
    { GemmParams p{}; p.A = bufC; p.lda = D_E;
      p.Bw = ffn_W1; p.ldb = D_FF;
      p.bias = ffn_b1; p.relu = 1;
      p.outH = ffh; p.ldoH = D_FF;
      p.M = D_TTOK; p.N = D_FF; p.K = D_E;
      launch_gemm(p, 1, 0, 1, stream); }

    // 9. FFN2 (+bias) -> f32 (reuses attnproj slot)
    { GemmParams p{}; p.A = ffh; p.lda = D_FF;
      p.Bw = ffn_W2; p.ldb = D_E;
      p.bias = ffn_b2;
      p.outF = scr1; p.ldoF = D_E;
      p.M = D_TTOK; p.N = D_E; p.K = D_FF;
      launch_gemm(p, 1, 0, 1, stream); }

    // 10. LN2(z + ffn_out) -> encoder output (reuses trainx slot)
    ln_kernel<<<D_TTOK / 8, 256, 0, stream>>>(zf, scr1, ln2_g, ln2_b, trainx, (half_t*)nullptr, D_TTOK);

    // 11. mean pool over train tokens
    pool_kernel<<<(D_B * D_E + 255) / 256, 256, 0, stream>>>(trainx, pooled);

    // 12. decoder hidden: dh = relu(pooled @ dec_W + dec_b) -> f16
    { GemmParams p{}; p.A = pooled; p.lda = D_E;
      p.Bw = dec_W; p.ldb = D_DEC;
      p.bias = dec_b; p.relu = 1;
      p.outH = dh_h; p.ldoH = D_DEC;
      p.M = D_B; p.N = D_DEC; p.K = D_E;
      launch_gemm(p, 0, 0, 1, stream); }

    // 13-16. generated MLP weights (M=16 skinny GEMMs; w1 is the 210MB stream)
    { GemmParams p{}; p.A = dh_h; p.lda = D_DEC;
      p.Bw = hw1; p.ldb = (long)D_F * D_H;
      p.outH = w1h; p.ldoH = (long)D_F * D_H;
      p.M = D_B; p.N = D_F * D_H; p.K = D_DEC;
      launch_gemm(p, 1, 0, 1, stream); }
    { GemmParams p{}; p.A = dh_h; p.lda = D_DEC;
      p.Bw = hb1; p.ldb = D_H;
      p.outF = b1f; p.ldoF = D_H;
      p.M = D_B; p.N = D_H; p.K = D_DEC;
      launch_gemm(p, 1, 0, 1, stream); }
    { GemmParams p{}; p.A = dh_h; p.lda = D_DEC;
      p.Bw = hw2; p.ldb = (long)D_H * D_NOUT;
      p.outH = w2h; p.ldoH = (long)D_H * D_NOUT;
      p.M = D_B; p.N = D_H * D_NOUT; p.K = D_DEC;
      launch_gemm(p, 1, 0, 1, stream); }
    { GemmParams p{}; p.A = dh_h; p.lda = D_DEC;
      p.Bw = hb2; p.ldb = D_NOUT;
      p.outF = b2f; p.ldoF = D_NOUT;
      p.M = D_B; p.N = D_NOUT; p.K = D_DEC;
      launch_gemm(p, 1, 0, 1, stream); }

    // 17. eval MLP layer 1 (batched over b via grid.z): h = relu(x_eval[:,b]@w1[b] + b1[b])
    { GemmParams p{}; p.A = x + (size_t)D_SEP * D_B * D_F;
      p.lda = (long)D_B * D_F; p.aBatch = D_F; p.sanA = 1;
      p.Bw = w1h; p.ldb = D_H; p.bBatch = (long)D_F * D_H;
      p.bias = b1f; p.biasBatch = D_H;
      p.relu = 1;
      p.outH = bufD; p.ldoH = (long)D_B * D_H; p.outHBatch = D_H;  // [e,b,h]
      p.M = D_EVAL; p.N = D_H; p.K = D_F;
      launch_gemm(p, 0, 1, D_B, stream); }

    // 18. eval MLP layer 2: out[e,b,:] = h[e,b]@w2[b] + b2[b]
    { GemmParams p{}; p.A = bufD;
      p.lda = (long)D_B * D_H; p.aBatch = D_H;
      p.Bw = w2h; p.ldb = D_NOUT; p.bBatch = (long)D_H * D_NOUT;
      p.bias = b2f; p.biasBatch = D_NOUT;
      p.outF = out; p.ldoF = (long)D_B * D_NOUT; p.outFBatch = D_NOUT;
      p.M = D_EVAL; p.N = D_NOUT; p.K = D_H;
      launch_gemm(p, 1, 1, D_B, stream); }
}
